// BinaryNet_31696858644686
// MI455X (gfx1250) — 8-bit matrix operand order repaired and checked by reading, compiled, not run on hardware
//
#include <hip/hip_runtime.h>
#include <hip/hip_bf16.h>
#include <stdint.h>

typedef __attribute__((ext_vector_type(16))) int   v16i;
typedef __attribute__((ext_vector_type(8)))  float v8f;

#define FP8_P1 0x38
#define FP8_M1 0xB8
#define FP8_Z  0x00
#define NB 128

static __device__ __forceinline__ float bn_thr(const float* g, const float* b,
                                               const float* m, const float* v, int c) {
    float inv = g[c] * rsqrtf(v[c] + 1e-5f);
    return m[c] - b[c] / inv;
}

__global__ void conv1_bn_bin_kernel(const float* __restrict__ x,
                                    const float* __restrict__ w,
                                    const float* __restrict__ g, const float* __restrict__ be,
                                    const float* __restrict__ mn, const float* __restrict__ vr,
                                    uint8_t* __restrict__ act)
{
    int idx = blockIdx.x * blockDim.x + threadIdx.x;
    if (idx >= NB * 34 * 34 * 128) return;
    int co = idx & 127;
    int p  = idx >> 7;
    int xi = p % 34; p /= 34;
    int yi = p % 34; int n = p / 34;
    uint8_t vb = FP8_Z;
    if (yi >= 1 && yi <= 32 && xi >= 1 && xi <= 32) {
        int yy = yi - 1, xx = xi - 1;
        float acc = 0.f;
        #pragma unroll
        for (int ci = 0; ci < 3; ++ci)
            #pragma unroll
            for (int dy = -1; dy <= 1; ++dy)
                #pragma unroll
                for (int dx = -1; dx <= 1; ++dx) {
                    int y2 = yy + dy, x2 = xx + dx;
                    if ((unsigned)y2 < 32u && (unsigned)x2 < 32u) {
                        float xv = x[((n * 3 + ci) * 32 + y2) * 32 + x2];
                        float wv = w[((co * 3 + ci) * 3 + (dy + 1)) * 3 + (dx + 1)];
                        acc = fmaf(xv, wv, acc);
                    }
                }
        vb = (acc >= bn_thr(g, be, mn, vr, co)) ? FP8_P1 : FP8_M1;
    }
    act[idx] = vb;
}

__global__ void pack_wbin_kernel(const float* __restrict__ w, uint8_t* __restrict__ pk,
                                 int C, int Cout)
{
    int total = Cout * 9 * C;
    int i = blockIdx.x * blockDim.x + threadIdx.x;
    if (i >= total) return;
    int kChunks = (9 * C) >> 7;
    int byte = i & 63;  int r = i >> 6;
    int lane = r & 31;  r >>= 5;
    int kc = r % kChunks;
    int ct = r / kChunks;
    int m = lane & 15, half = lane >> 4;
    int v = byte >> 2, b = byte & 3;
    int hi = v >> 3, vv = v & 7, gg = vv >> 1, d = vv & 1;
    int kl = (v >> 2) * 32 + half * 16 + (v & 3) * 4 + b;
    int K  = (kc << 7) + kl;
    int t  = K / C, c = K % C;
    int dy = t / 3, dx = t % 3;
    int co = ct * 16 + m;
    float wv = w[((co * C + c) * 3 + dy) * 3 + dx];
    pk[i] = (wv >= 0.f) ? FP8_P1 : FP8_M1;
}

template<int H, int W, int C, int Cout>
__global__ __launch_bounds__(256)
void bconv_gemm_fp8_kernel(const uint8_t* __restrict__ act,
                           const uint8_t* __restrict__ wpk,
                           float* __restrict__ out)
{
    constexpr int PW  = W + 2, PH = H + 2;
    constexpr int KCH = (9 * C) / 128;
    constexpr int CCH = C / 128;
    const int lane = threadIdx.x & 31;
    const int wv   = threadIdx.x >> 5;
    const int col  = lane & 15;
    const int half = lane >> 4;

    const int pixBase = blockIdx.x * 512 + wv * 64 + col;
    const int ct0 = blockIdx.y * 2;

    const uint8_t* ctr[4];
    #pragma unroll
    for (int j = 0; j < 4; ++j) {
        int p = pixBase + 16 * j;
        int n = p / (H * W);
        int r = p & (H * W - 1);
        int y = r / W, x = r & (W - 1);
        ctr[j] = act + ((size_t)(n * PH + y + 1) * PW + (x + 1)) * C + half * 16;
    }

    v8f acc[2][4] = {};
    const uint8_t* aLane = wpk + (size_t)ct0 * (KCH * 2048) + (size_t)lane * 64;

    #pragma unroll
    for (int t = 0; t < 9; ++t) {
        const int dy = t / 3 - 1, dx = t % 3 - 1;
        const int tapOff = (dy * PW + dx) * C;
        #pragma unroll
        for (int cc = 0; cc < CCH; ++cc) {
            const int kc = t * CCH + cc;
            const v16i a0 = *(const v16i*)(aLane + kc * 2048);
            const v16i a1 = *(const v16i*)(aLane + kc * 2048 + KCH * 2048);
            v16i b[4];
            #pragma unroll
            for (int j = 0; j < 4; ++j) {
                union { int4 q[4]; v16i v; } u;
                #pragma unroll
                for (int g2 = 0; g2 < 4; ++g2)
                    u.q[g2] = *(const int4*)(ctr[j] + tapOff + cc * 128 + g2 * 32);
                b[j] = u.v;
            }
            #pragma unroll
            for (int j = 0; j < 4; ++j) {
                acc[0][j] = __builtin_amdgcn_wmma_f32_16x16x128_fp8_fp8(
                    a0, b[j], (short)0, acc[0][j], false, false);
                acc[1][j] = __builtin_amdgcn_wmma_f32_16x16x128_fp8_fp8(
                    a1, b[j], (short)0, acc[1][j], false, false);
            }
        }
    }

    #pragma unroll
    for (int i = 0; i < 2; ++i)
        #pragma unroll
        for (int j = 0; j < 4; ++j) {
            const int p  = pixBase + 16 * j;
            const int co = (ct0 + i) * 16 + half * 8;
            float* dst = out + (size_t)p * Cout + co;
            *(float4*)(dst)     = make_float4(acc[i][j][0], acc[i][j][1],
                                              acc[i][j][2], acc[i][j][3]);
            *(float4*)(dst + 4) = make_float4(acc[i][j][4], acc[i][j][5],
                                              acc[i][j][6], acc[i][j][7]);
        }
}

__global__ void thresh_pad_kernel(const float* __restrict__ in,
                                  uint8_t* __restrict__ outp,
                                  const float* __restrict__ g, const float* __restrict__ be,
                                  const float* __restrict__ mn, const float* __restrict__ vr,
                                  int H, int W, int C, int total)
{
    int i = blockIdx.x * blockDim.x + threadIdx.x;
    if (i >= total) return;
    int c = i & (C - 1);
    int p = i / C;
    int PW = W + 2, PH = H + 2;
    int xi = p % PW; p /= PW;
    int yi = p % PH; int n = p / PH;
    uint8_t v = FP8_Z;
    if (yi >= 1 && yi <= H && xi >= 1 && xi <= W) {
        float val = in[((size_t)(n * H + yi - 1) * W + (xi - 1)) * C + c];
        v = (val >= bn_thr(g, be, mn, vr, c)) ? FP8_P1 : FP8_M1;
    }
    outp[i] = v;
}

__global__ void pool_thresh_pad_kernel(const float* __restrict__ in,
                                       uint8_t* __restrict__ outp,
                                       const float* __restrict__ g, const float* __restrict__ be,
                                       const float* __restrict__ mn, const float* __restrict__ vr,
                                       int H, int W, int C, int total)
{
    int i = blockIdx.x * blockDim.x + threadIdx.x;
    if (i >= total) return;
    int c = i & (C - 1);
    int p = i / C;
    int Ho = H >> 1, Wo = W >> 1;
    int PW = Wo + 2, PH = Ho + 2;
    int xi = p % PW; p /= PW;
    int yi = p % PH; int n = p / PH;
    uint8_t v = FP8_Z;
    if (yi >= 1 && yi <= Ho && xi >= 1 && xi <= Wo) {
        int ys = 2 * (yi - 1), xs = 2 * (xi - 1);
        const float* b0 = in + ((size_t)(n * H + ys) * W + xs) * C + c;
        float mx = fmaxf(fmaxf(b0[0], b0[C]),
                         fmaxf(b0[(size_t)W * C], b0[(size_t)W * C + C]));
        v = (mx >= bn_thr(g, be, mn, vr, c)) ? FP8_P1 : FP8_M1;
    }
    outp[i] = v;
}

__global__ void pool_bn_ht_kernel(const float* __restrict__ in,
                                  float* __restrict__ feat,
                                  const float* __restrict__ g, const float* __restrict__ be,
                                  const float* __restrict__ mn, const float* __restrict__ vr)
{
    int i = blockIdx.x * blockDim.x + threadIdx.x;
    if (i >= NB * 512 * 16) return;
    int xo = i & 3, yo = (i >> 2) & 3, c = (i >> 4) & 511, n = i >> 13;
    const float* b0 = in + (size_t)((n * 8 + 2 * yo) * 8 + 2 * xo) * 512 + c;
    float mx = fmaxf(fmaxf(b0[0], b0[512]), fmaxf(b0[8 * 512], b0[8 * 512 + 512]));
    float inv = g[c] * rsqrtf(vr[c] + 1e-5f);
    float val = mx * inv + (be[c] - mn[c] * inv);
    feat[i] = fminf(1.f, fmaxf(-1.f, val));
}

__global__ void fc_kernel(const float* __restrict__ feat, const float* __restrict__ wfc,
                          const float* __restrict__ bfc, float* __restrict__ outp)
{
    int gtid = blockIdx.x * blockDim.x + threadIdx.x;
    int wid  = gtid >> 5;
    int lane = gtid & 31;
    if (wid >= NB * 100) return;
    int j = wid % 100, n = wid / 100;
    const float* f  = feat + (size_t)n * 8192;
    const float* wr = wfc  + (size_t)j * 8192;
    float s = 0.f;
    #pragma unroll 4
    for (int k = lane; k < 8192; k += 32) s = fmaf(f[k], wr[k], s);
    #pragma unroll
    for (int off = 16; off > 0; off >>= 1) s += __shfl_down(s, off, 32);
    if (lane == 0) outp[wid] = bfc[j] + s;
}

extern "C" void kernel_launch(void* const* d_in, const int* in_sizes, int n_in,
                              void* d_out, int out_size, void* d_ws, size_t ws_size,
                              hipStream_t stream) {
    (void)in_sizes; (void)n_in; (void)out_size; (void)ws_size;
    const float* x = (const float*)d_in[0];
    const float *cw[6], *bg[6], *bb[6], *bm[6], *bv[6];
    for (int i = 0; i < 6; ++i) {
        cw[i] = (const float*)d_in[1 + i];
        bg[i] = (const float*)d_in[7 + i];
        bb[i] = (const float*)d_in[13 + i];
        bm[i] = (const float*)d_in[19 + i];
        bv[i] = (const float*)d_in[25 + i];
    }
    const float* fcw = (const float*)d_in[31];
    const float* fcb = (const float*)d_in[32];
    float* out = (float*)d_out;

    uint8_t* ws      = (uint8_t*)d_ws;
    float*   convOut = (float*)ws;
    uint8_t* actA    = ws + (size_t)(64u << 20);
    uint8_t* actB    = actA + (size_t)(20u << 20);
    uint8_t* wpBase  = actB + (size_t)(20u << 20);

    static const int Cl[5]  = {128, 128, 256, 256, 512};
    static const int Col[5] = {128, 256, 256, 512, 512};

    size_t wOff[5]; size_t acc = 0;
    for (int i = 0; i < 5; ++i) { wOff[i] = acc; acc += (size_t)Col[i] * 9 * Cl[i]; }
    float* feat = (float*)(wpBase + ((acc + 255) & ~(size_t)255));

    for (int i = 0; i < 5; ++i) {
        int total = Col[i] * 9 * Cl[i];
        pack_wbin_kernel<<<(total + 255) / 256, 256, 0, stream>>>(
            cw[i + 1], wpBase + wOff[i], Cl[i], Col[i]);
    }

    {
        int total = NB * 34 * 34 * 128;
        conv1_bn_bin_kernel<<<(total + 255) / 256, 256, 0, stream>>>(
            x, cw[0], bg[0], bb[0], bm[0], bv[0], actA);
    }

    bconv_gemm_fp8_kernel<32, 32, 128, 128>
        <<<dim3(NB * 32 * 32 / 512, 128 / 32), 256, 0, stream>>>(actA, wpBase + wOff[0], convOut);
    {
        int total = NB * 18 * 18 * 128;
        pool_thresh_pad_kernel<<<(total + 255) / 256, 256, 0, stream>>>(
            convOut, actB, bg[1], bb[1], bm[1], bv[1], 32, 32, 128, total);
    }

    bconv_gemm_fp8_kernel<16, 16, 128, 256>
        <<<dim3(NB * 16 * 16 / 512, 256 / 32), 256, 0, stream>>>(actB, wpBase + wOff[1], convOut);
    {
        int total = NB * 18 * 18 * 256;
        thresh_pad_kernel<<<(total + 255) / 256, 256, 0, stream>>>(
            convOut, actA, bg[2], bb[2], bm[2], bv[2], 16, 16, 256, total);
    }

    bconv_gemm_fp8_kernel<16, 16, 256, 256>
        <<<dim3(NB * 16 * 16 / 512, 256 / 32), 256, 0, stream>>>(actA, wpBase + wOff[2], convOut);
    {
        int total = NB * 10 * 10 * 256;
        pool_thresh_pad_kernel<<<(total + 255) / 256, 256, 0, stream>>>(
            convOut, actB, bg[3], bb[3], bm[3], bv[3], 16, 16, 256, total);
    }

    bconv_gemm_fp8_kernel<8, 8, 256, 512>
        <<<dim3(NB * 8 * 8 / 512, 512 / 32), 256, 0, stream>>>(actB, wpBase + wOff[3], convOut);
    {
        int total = NB * 10 * 10 * 512;
        thresh_pad_kernel<<<(total + 255) / 256, 256, 0, stream>>>(
            convOut, actA, bg[4], bb[4], bm[4], bv[4], 8, 8, 512, total);
    }

    bconv_gemm_fp8_kernel<8, 8, 512, 512>
        <<<dim3(NB * 8 * 8 / 512, 512 / 32), 256, 0, stream>>>(actA, wpBase + wOff[4], convOut);
    pool_bn_ht_kernel<<<(NB * 512 * 16) / 256, 256, 0, stream>>>(
        convOut, feat, bg[5], bb[5], bm[5], bv[5]);

    fc_kernel<<<(NB * 100 * 32 + 255) / 256, 256, 0, stream>>>(feat, fcw, fcb, out);
}
